// ImprovedGNNModel_5385888989827
// MI455X (gfx1250) — matrix operand order repaired and checked by reading, not compiled, not run on hardware
//
#include <hip/hip_runtime.h>
#include <math.h>

// ======================================================================
// ImprovedGNNModel for MI455X (gfx1250, wave32, WMMA)
//
// Dense linears  -> v_wmma_f32_16x16x32_bf16, K templated (16/128) so the
//                   K-loop fully unrolls: unconditional float4 A loads,
//                   branchless B loads, 4 back-to-back WMMAs for K=128.
// Edge phase     -> wave-per-edge float4 gathers (node rows are L2-resident:
//                   192MB L2 >> 82MB working set), segment softmax via
//                   ordered-uint atomicMax + f32 atomicAdd scatter.
// ======================================================================

#define NN   40000   // nodes
#define NE   640000  // edges
#define NG   64      // graphs
#define FIN  16
#define HC   128     // heads*channels

// ---------- input index map ----------
// Top-level dict insertion order: x, edge_index, edge_attr, batch, params.
// params flattened as a JAX pytree: dict keys sorted alphabetically at each
// level.  params: bn1,bn2,bn3,conv1,conv2,conv3,fc,gate
//   bnX   : beta,gamma,mean,var
//   convX : e{W}, k{W,b}, q{W,b}, s{W,b}, v{W,b}
//   fc    : bn1{...}, bn2{...}, l1{W,b}, l2{W,b}, l3{W,b}
//   gate  : l1{W,b}, l2{W,b}
enum {
  IN_X = 0, IN_EI, IN_EA, IN_BATCH,
  BN1_BETA, BN1_GAMMA, BN1_MEAN, BN1_VAR,
  BN2_BETA, BN2_GAMMA, BN2_MEAN, BN2_VAR,
  BN3_BETA, BN3_GAMMA, BN3_MEAN, BN3_VAR,
  C1_EW, C1_KW, C1_KB, C1_QW, C1_QB, C1_SW, C1_SB, C1_VW, C1_VB,
  C2_EW, C2_KW, C2_KB, C2_QW, C2_QB, C2_SW, C2_SB, C2_VW, C2_VB,
  C3_EW, C3_KW, C3_KB, C3_QW, C3_QB, C3_SW, C3_SB, C3_VW, C3_VB,
  FCBN1_BETA, FCBN1_GAMMA, FCBN1_MEAN, FCBN1_VAR,
  FCBN2_BETA, FCBN2_GAMMA, FCBN2_MEAN, FCBN2_VAR,
  FC1_W, FC1_B, FC2_W, FC2_B, FC3_W, FC3_B,
  G1_W, G1_B, G2_W, G2_B
};

typedef __attribute__((ext_vector_type(16))) __bf16 v16bf;
typedef __attribute__((ext_vector_type(8)))  float  v8f;

// ---- monotone float <-> ordered-uint encoding for atomic float max ----
__device__ __forceinline__ unsigned f2ord(float f) {
  unsigned u = __float_as_uint(f);
  return (u & 0x80000000u) ? ~u : (u | 0x80000000u);
}
__device__ __forceinline__ float ord2f(unsigned u) {
  return __uint_as_float((u & 0x80000000u) ? (u & 0x7fffffffu) : ~u);
}
#define ORD_NEG_INF 0x007FFFFFu   // f2ord(-inf)

// ======================================================================
// WMMA GEMM:  Y[M,Nd] = X[M,KD] @ W[KD,Nd] (+ bias) (+ relu)
// One wave computes one 16x16 tile. bf16 A/B, f32 accumulate. KD is a
// compile-time constant (16 or 128) so the K loop fully unrolls and all
// range checks fold away -> unconditional vector loads, no EXEC churn.
//
// A layout (16-bit A 16x32, ISA 7.12.2): lane row m=L%16, half=L/16,
//   elem e<8 -> K = kk + half*8 + e ; e>=8 -> K = kk + 16 + half*8 + (e-8)
//   => two contiguous 8-float runs per lane: 4x float4 loads.
// B layout: lane col n=L%16, half=L/16, same K mapping as A (table order)
// D layout: lane L, acc[r] -> row tM*16 + r + 8*half, col tN*16 + L%16
// EPI: 1 = +bias, 2 = +bias+relu
// ======================================================================
template<int KD, int EPI>
__global__ __launch_bounds__(128)
void gemm_bf16_wmma_kernel(const float* __restrict__ X, const float* __restrict__ W,
                           const float* __restrict__ bias, float* __restrict__ Y,
                           int M, int Nd) {
  int wave = blockIdx.x * (blockDim.x >> 5) + (threadIdx.x >> 5);
  int tilesN = Nd >> 4;
  int tilesTotal = (M >> 4) * tilesN;
  if (wave >= tilesTotal) return;
  int tM = wave / tilesN;
  int tN = wave - tM * tilesN;
  int lane = threadIdx.x & 31;
  int half = lane >> 4;
  int idx16 = lane & 15;
  int arow = tM * 16 + idx16;   // row this lane feeds for A
  int bcol = tN * 16 + idx16;   // col this lane feeds for B
  const float* __restrict__ xrow = X + (size_t)arow * KD;

  v8f acc = {};
#pragma unroll
  for (int kk = 0; kk < KD; kk += 32) {
    // ---- A: unconditional float4 loads (32B aligned) ----
    float4 a0 = *(const float4*)(xrow + kk + half * 8);
    float4 a1 = *(const float4*)(xrow + kk + half * 8 + 4);
    float4 a2, a3;
    if constexpr (KD % 32 == 0) {
      a2 = *(const float4*)(xrow + kk + 16 + half * 8);
      a3 = *(const float4*)(xrow + kk + 16 + half * 8 + 4);
    } else {                       // KD==16: K in [16,32) doesn't exist
      a2 = make_float4(0.f, 0.f, 0.f, 0.f);
      a3 = make_float4(0.f, 0.f, 0.f, 0.f);
    }
    v16bf a, b;
    a[0]  = (__bf16)a0.x; a[1]  = (__bf16)a0.y; a[2]  = (__bf16)a0.z; a[3]  = (__bf16)a0.w;
    a[4]  = (__bf16)a1.x; a[5]  = (__bf16)a1.y; a[6]  = (__bf16)a1.z; a[7]  = (__bf16)a1.w;
    a[8]  = (__bf16)a2.x; a[9]  = (__bf16)a2.y; a[10] = (__bf16)a2.z; a[11] = (__bf16)a2.w;
    a[12] = (__bf16)a3.x; a[13] = (__bf16)a3.y; a[14] = (__bf16)a3.z; a[15] = (__bf16)a3.w;

    // ---- B: strided column loads, branchless ----
    if constexpr (KD % 32 == 0) {
#pragma unroll
      for (int e = 0; e < 16; ++e)
        b[e] = (__bf16)W[(size_t)(kk + half * 8 + e + (e >= 8 ? 8 : 0)) * Nd + bcol];
    } else {
      // KD==16: elements e>=8 of both lane halves address rows >= KD; clamp the address
      // (always in-bounds) and cndmask the value to zero -> no branches.
#pragma unroll
      for (int e = 0; e < 16; ++e) {
        int kb = kk + half * 8 + e + (e >= 8 ? 8 : 0);
        int kcl = kb < KD ? kb : KD - 1;
        float wv = W[(size_t)kcl * Nd + bcol];
        b[e] = (__bf16)(kb < KD ? wv : 0.0f);
      }
    }
    acc = __builtin_amdgcn_wmma_f32_16x16x32_bf16(false, a, false, b,
                                                  (short)0, acc, false, false);
  }
  float bv = bias[bcol];
#pragma unroll
  for (int r = 0; r < 8; ++r) {
    int orow = tM * 16 + (half << 3) + r;
    float v = acc[r] + bv;
    if (EPI == 2) v = fmaxf(v, 0.0f);
    Y[(size_t)orow * Nd + bcol] = v;
  }
}

// ======================================================================
// init per-node attention state: amax = -inf (encoded), denom = 0
// ======================================================================
__global__ void init_node_kernel(unsigned* __restrict__ amax, float* __restrict__ den, int n) {
  int t = blockIdx.x * blockDim.x + threadIdx.x;
  if (t >= n) return;
  amax[t] = ORD_NEG_INF;
  den[t] = 0.0f;
}

// ======================================================================
// edge pass 1: alpha[e,h] = <Q[dst,h,:], K[src,h,:]+Eproj[h,:]> / 8
//              atomicMax amax[dst,h]
// One wave per edge: 4 channels/lane; lanes 0-15 = head0, 16-31 = head1.
// ======================================================================
__global__ void edge_alpha_kernel(const float* __restrict__ Q, const float* __restrict__ K,
                                  const float* __restrict__ eW, const float* __restrict__ ea,
                                  const int* __restrict__ src, const int* __restrict__ dst,
                                  float* __restrict__ alpha, unsigned* __restrict__ amax,
                                  int nE) {
  int wid = (blockIdx.x * blockDim.x + threadIdx.x) >> 5;
  if (wid >= nE) return;
  int lane = threadIdx.x & 31;
  int c = lane * 4;
  int s = src[wid], d = dst[wid];
  float4 q = *(const float4*)(Q + (size_t)d * HC + c);
  float4 k = *(const float4*)(K + (size_t)s * HC + c);
  float e0 = ea[(size_t)wid * 4 + 0], e1 = ea[(size_t)wid * 4 + 1];
  float e2 = ea[(size_t)wid * 4 + 2], e3 = ea[(size_t)wid * 4 + 3];
  float4 w0 = *(const float4*)(eW + c);
  float4 w1 = *(const float4*)(eW + HC + c);
  float4 w2 = *(const float4*)(eW + 2 * HC + c);
  float4 w3 = *(const float4*)(eW + 3 * HC + c);
  float4 ep;
  ep.x = e0 * w0.x + e1 * w1.x + e2 * w2.x + e3 * w3.x;
  ep.y = e0 * w0.y + e1 * w1.y + e2 * w2.y + e3 * w3.y;
  ep.z = e0 * w0.z + e1 * w1.z + e2 * w2.z + e3 * w3.z;
  ep.w = e0 * w0.w + e1 * w1.w + e2 * w2.w + e3 * w3.w;
  float p = q.x * (k.x + ep.x) + q.y * (k.y + ep.y) + q.z * (k.z + ep.z) + q.w * (k.w + ep.w);
  // reduce 16 lanes per head half (wave32)
  p += __shfl_xor(p, 8, 32);
  p += __shfl_xor(p, 4, 32);
  p += __shfl_xor(p, 2, 32);
  p += __shfl_xor(p, 1, 32);
  if ((lane & 15) == 0) {
    int h = lane >> 4;
    float a = p * 0.125f;                       // 1/sqrt(C=64)
    alpha[(size_t)wid * 2 + h] = a;
    atomicMax(&amax[(size_t)d * 2 + h], f2ord(a));
  }
}

// ======================================================================
// edge pass 2: ex = exp(alpha - amax[dst]); denom[dst] += ex  (in place)
// ======================================================================
__global__ void edge_exp_kernel(float* __restrict__ alpha, const unsigned* __restrict__ amax,
                                const int* __restrict__ dst, float* __restrict__ den, int nEH) {
  int t = blockIdx.x * blockDim.x + threadIdx.x;
  if (t >= nEH) return;
  int e = t >> 1, h = t & 1;
  int d = dst[e];
  float x = __expf(alpha[t] - ord2f(amax[(size_t)d * 2 + h]));
  alpha[t] = x;
  atomicAdd(&den[(size_t)d * 2 + h], x);
}

// ======================================================================
// edge pass 3: out[dst,:] += (V[src,:]+Eproj) * ex/(den[dst]+eps)
// ======================================================================
__global__ void edge_scatter_kernel(const float* __restrict__ V, const float* __restrict__ eW,
                                    const float* __restrict__ ea,
                                    const int* __restrict__ src, const int* __restrict__ dst,
                                    const float* __restrict__ ex, const float* __restrict__ den,
                                    float* __restrict__ out, int nE) {
  int wid = (blockIdx.x * blockDim.x + threadIdx.x) >> 5;
  if (wid >= nE) return;
  int lane = threadIdx.x & 31;
  int h = lane >> 4;
  int c = lane * 4;
  int s = src[wid], d = dst[wid];
  float w = ex[(size_t)wid * 2 + h] / (den[(size_t)d * 2 + h] + 1e-16f);
  float4 v = *(const float4*)(V + (size_t)s * HC + c);
  float e0 = ea[(size_t)wid * 4 + 0], e1 = ea[(size_t)wid * 4 + 1];
  float e2 = ea[(size_t)wid * 4 + 2], e3 = ea[(size_t)wid * 4 + 3];
  float4 w0 = *(const float4*)(eW + c);
  float4 w1 = *(const float4*)(eW + HC + c);
  float4 w2 = *(const float4*)(eW + 2 * HC + c);
  float4 w3 = *(const float4*)(eW + 3 * HC + c);
  float4 ep;
  ep.x = e0 * w0.x + e1 * w1.x + e2 * w2.x + e3 * w3.x;
  ep.y = e0 * w0.y + e1 * w1.y + e2 * w2.y + e3 * w3.y;
  ep.z = e0 * w0.z + e1 * w1.z + e2 * w2.z + e3 * w3.z;
  ep.w = e0 * w0.w + e1 * w1.w + e2 * w2.w + e3 * w3.w;
  float* o = out + (size_t)d * HC + c;
  atomicAdd(o + 0, (v.x + ep.x) * w);
  atomicAdd(o + 1, (v.y + ep.y) * w);
  atomicAdd(o + 2, (v.z + ep.z) * w);
  atomicAdd(o + 3, (v.w + ep.w) * w);
}

// ======================================================================
// eval-mode BN + ReLU (+ optional residual added AFTER relu, matching ref)
// ======================================================================
__global__ void bn_relu_kernel(const float* __restrict__ in, const float* __restrict__ gamma,
                               const float* __restrict__ beta, const float* __restrict__ mean,
                               const float* __restrict__ var, const float* __restrict__ res,
                               float* __restrict__ out, int total, int Cd) {
  int t = blockIdx.x * blockDim.x + threadIdx.x;
  if (t >= total) return;
  int c = t % Cd;
  float sc = gamma[c] * rsqrtf(var[c] + 1e-5f);
  float y = (in[t] - mean[c]) * sc + beta[c];
  y = fmaxf(y, 0.0f);
  if (res) y += res[t];
  out[t] = y;
}

// y[i] = b + sum_{j<64} X[i*64+j] * w[j]
__global__ void rowdot64_kernel(const float* __restrict__ X, const float* __restrict__ w,
                                const float* __restrict__ b, float* __restrict__ y, int n) {
  int i = blockIdx.x * blockDim.x + threadIdx.x;
  if (i >= n) return;
  float acc = b[0];
#pragma unroll
  for (int j = 0; j < 64; ++j) acc += X[(size_t)i * 64 + j] * w[j];
  y[i] = acc;
}

// ---------------- global-attention pooling ----------------
__global__ void pool_init_kernel(unsigned* __restrict__ gmax, float* __restrict__ gden,
                                 float* __restrict__ pool) {
  int t = blockIdx.x * blockDim.x + threadIdx.x;
  if (t < NG) { gmax[t] = ORD_NEG_INF; gden[t] = 0.0f; }
  if (t < NG * HC) pool[t] = 0.0f;
}
__global__ void pool_max_kernel(const float* __restrict__ g, const int* __restrict__ batch,
                                unsigned* __restrict__ gmax, int n) {
  int t = blockIdx.x * blockDim.x + threadIdx.x;
  if (t >= n) return;
  atomicMax(&gmax[batch[t]], f2ord(g[t]));
}
__global__ void pool_exp_kernel(const float* __restrict__ g, const int* __restrict__ batch,
                                const unsigned* __restrict__ gmax, float* __restrict__ wg,
                                float* __restrict__ gden, int n) {
  int t = blockIdx.x * blockDim.x + threadIdx.x;
  if (t >= n) return;
  int b = batch[t];
  float x = __expf(g[t] - ord2f(gmax[b]));
  wg[t] = x;
  atomicAdd(&gden[b], x);
}
__global__ void pool_acc_kernel(const float* __restrict__ h, const float* __restrict__ wg,
                                const float* __restrict__ gden, const int* __restrict__ batch,
                                float* __restrict__ pool, int n) {
  int t = blockIdx.x * blockDim.x + threadIdx.x;
  if (t >= n * HC) return;
  int i = t >> 7, c = t & (HC - 1);
  int b = batch[i];
  float w = wg[i] / (gden[b] + 1e-16f);
  atomicAdd(&pool[(size_t)b * HC + c], h[t] * w);
}

// ======================================================================
// host side
// ======================================================================
static inline int cdiv(int a, int b) { return (a + b - 1) / b; }

template<int KD, int EPI>
static inline void gemm(const float* X, const float* W, const float* B, float* Y,
                        int M, int Nd, hipStream_t s) {
  int tiles = (M / 16) * (Nd / 16);
  gemm_bf16_wmma_kernel<KD, EPI><<<cdiv(tiles, 4), 128, 0, s>>>(X, W, B, Y, M, Nd);
}

struct ConvW {
  const float *eW, *kW, *kB, *qW, *qB, *sW, *sB, *vW, *vB;
  const float *bnBeta, *bnGamma, *bnMean, *bnVar;
};

// One TransformerConv layer + BN + ReLU (+residual).  Qb doubles as the
// output accumulator (root/skip gemm then message scatter), Kb doubles as V.
template<int DIN>
static void run_conv(const ConvW& w, const float* hin, const float* res,
                     float* hout, float* Qb, float* Kb, float* ALPHA, unsigned* AMAX,
                     float* DEN, const int* src, const int* dst, const float* ea,
                     hipStream_t s) {
  gemm<DIN, 1>(hin, w.qW, w.qB, Qb, NN, HC, s);                 // Q
  gemm<DIN, 1>(hin, w.kW, w.kB, Kb, NN, HC, s);                 // K
  init_node_kernel<<<cdiv(NN * 2, 256), 256, 0, s>>>(AMAX, DEN, NN * 2);
  edge_alpha_kernel<<<cdiv(NE * 32, 256), 256, 0, s>>>(Qb, Kb, w.eW, ea, src, dst,
                                                       ALPHA, AMAX, NE);
  gemm<DIN, 1>(hin, w.vW, w.vB, Kb, NN, HC, s);                 // V (K dead)
  gemm<DIN, 1>(hin, w.sW, w.sB, Qb, NN, HC, s);                 // skip -> out acc
  edge_exp_kernel<<<cdiv(NE * 2, 256), 256, 0, s>>>(ALPHA, AMAX, dst, DEN, NE * 2);
  edge_scatter_kernel<<<cdiv(NE * 32, 256), 256, 0, s>>>(Kb, w.eW, ea, src, dst,
                                                         ALPHA, DEN, Qb, NE);
  bn_relu_kernel<<<cdiv(NN * HC, 256), 256, 0, s>>>(Qb, w.bnGamma, w.bnBeta, w.bnMean,
                                                    w.bnVar, res, hout, NN * HC, HC);
}

extern "C" void kernel_launch(void* const* d_in, const int* in_sizes, int n_in,
                              void* d_out, int out_size, void* d_ws, size_t ws_size,
                              hipStream_t stream) {
  const float* x     = (const float*)d_in[IN_X];
  const int*   ei    = (const int*)d_in[IN_EI];
  const float* ea    = (const float*)d_in[IN_EA];
  const int*   batch = (const int*)d_in[IN_BATCH];
  const int* src = ei;
  const int* dst = ei + NE;
  float* out = (float*)d_out;

  // ---- workspace carving (floats; all 4B aligned) ----
  float* ws = (float*)d_ws;
  float*    Qb    = ws;                          // N*HC  (Q, then out acc)
  float*    Kb    = Qb + (size_t)NN * HC;        // N*HC  (K, then V, then gate hidden)
  float*    HA    = Kb + (size_t)NN * HC;        // N*HC
  float*    HB    = HA + (size_t)NN * HC;        // N*HC
  float*    ALPHA = HB + (size_t)NN * HC;        // E*2
  unsigned* AMAX  = (unsigned*)(ALPHA + (size_t)NE * 2);  // N*2
  float*    DEN   = (float*)(AMAX + (size_t)NN * 2);      // N*2
  float*    GSC   = DEN + (size_t)NN * 2;        // N   gate scores
  float*    WG    = GSC + NN;                    // N   softmax numerators
  float*    POOL  = WG + NN;                     // G*HC
  unsigned* GMAX  = (unsigned*)(POOL + NG * HC); // G
  float*    GDEN  = (float*)(GMAX + NG);         // G
  float*    Z1    = GDEN + NG;                   // G*128
  float*    Z2    = Z1 + NG * 128;               // G*64

  ConvW c1 = {(const float*)d_in[C1_EW], (const float*)d_in[C1_KW], (const float*)d_in[C1_KB],
              (const float*)d_in[C1_QW], (const float*)d_in[C1_QB], (const float*)d_in[C1_SW],
              (const float*)d_in[C1_SB], (const float*)d_in[C1_VW], (const float*)d_in[C1_VB],
              (const float*)d_in[BN1_BETA], (const float*)d_in[BN1_GAMMA],
              (const float*)d_in[BN1_MEAN], (const float*)d_in[BN1_VAR]};
  ConvW c2 = {(const float*)d_in[C2_EW], (const float*)d_in[C2_KW], (const float*)d_in[C2_KB],
              (const float*)d_in[C2_QW], (const float*)d_in[C2_QB], (const float*)d_in[C2_SW],
              (const float*)d_in[C2_SB], (const float*)d_in[C2_VW], (const float*)d_in[C2_VB],
              (const float*)d_in[BN2_BETA], (const float*)d_in[BN2_GAMMA],
              (const float*)d_in[BN2_MEAN], (const float*)d_in[BN2_VAR]};
  ConvW c3 = {(const float*)d_in[C3_EW], (const float*)d_in[C3_KW], (const float*)d_in[C3_KB],
              (const float*)d_in[C3_QW], (const float*)d_in[C3_QB], (const float*)d_in[C3_SW],
              (const float*)d_in[C3_SB], (const float*)d_in[C3_VW], (const float*)d_in[C3_VB],
              (const float*)d_in[BN3_BETA], (const float*)d_in[BN3_GAMMA],
              (const float*)d_in[BN3_MEAN], (const float*)d_in[BN3_VAR]};

  // ---- 3 TransformerConv layers ----
  run_conv<FIN>(c1, x,  nullptr, HA, Qb, Kb, ALPHA, AMAX, DEN, src, dst, ea, stream);
  run_conv<HC >(c2, HA, HA,      HB, Qb, Kb, ALPHA, AMAX, DEN, src, dst, ea, stream);
  run_conv<HC >(c3, HB, HB,      HA, Qb, Kb, ALPHA, AMAX, DEN, src, dst, ea, stream);
  // final node embedding h = HA

  // ---- GlobalAttention pooling ----
  gemm<HC, 2>(HA, (const float*)d_in[G1_W], (const float*)d_in[G1_B], Kb, NN, 64, stream);
  rowdot64_kernel<<<cdiv(NN, 256), 256, 0, stream>>>(Kb, (const float*)d_in[G2_W],
                                                     (const float*)d_in[G2_B], GSC, NN);
  pool_init_kernel<<<cdiv(NG * HC, 256), 256, 0, stream>>>(GMAX, GDEN, POOL);
  pool_max_kernel<<<cdiv(NN, 256), 256, 0, stream>>>(GSC, batch, GMAX, NN);
  pool_exp_kernel<<<cdiv(NN, 256), 256, 0, stream>>>(GSC, batch, GMAX, WG, GDEN, NN);
  pool_acc_kernel<<<cdiv(NN * HC, 256), 256, 0, stream>>>(HA, WG, GDEN, batch, POOL, NN);

  // ---- MLP head ----
  gemm<HC, 1>(POOL, (const float*)d_in[FC1_W], (const float*)d_in[FC1_B], Z1, NG, 128, stream);
  bn_relu_kernel<<<cdiv(NG * 128, 256), 256, 0, stream>>>(
      Z1, (const float*)d_in[FCBN1_GAMMA], (const float*)d_in[FCBN1_BETA],
      (const float*)d_in[FCBN1_MEAN], (const float*)d_in[FCBN1_VAR], nullptr, Z1, NG * 128, 128);
  gemm<HC, 1>(Z1, (const float*)d_in[FC2_W], (const float*)d_in[FC2_B], Z2, NG, 64, stream);
  bn_relu_kernel<<<cdiv(NG * 64, 256), 256, 0, stream>>>(
      Z2, (const float*)d_in[FCBN2_GAMMA], (const float*)d_in[FCBN2_BETA],
      (const float*)d_in[FCBN2_MEAN], (const float*)d_in[FCBN2_VAR], nullptr, Z2, NG * 64, 64);
  rowdot64_kernel<<<1, 64, 0, stream>>>(Z2, (const float*)d_in[FC3_W],
                                        (const float*)d_in[FC3_B], out, NG);
}
